// TemporalMambaStack_21784074125466
// MI455X (gfx1250) — matrix operand order repaired and checked by reading, not compiled, not run on hardware
//
#include <hip/hip_runtime.h>

// ---------------------------------------------------------------- constants
#define DIM      512
#define N_LAYERS 2
#define D_STATE  16
#define D_CONV   4
#define D_INNER  1024          // EXPAND * DIM
#define DT_RANK  32            // (DIM + 15) / 16
#define BATCH    2
#define SEQ      2048
#define MROWS    (BATCH * SEQ) // 4096
#define KSTEP    32

typedef __attribute__((ext_vector_type(16))) __bf16          v16bf;
typedef __attribute__((ext_vector_type(8)))  float           v8f;
typedef __attribute__((ext_vector_type(8)))  unsigned short  ushort8;
typedef __attribute__((ext_vector_type(16))) unsigned short  ushort16;

// ------------------------------------------------------------- bf16 helpers
__device__ __forceinline__ unsigned short f2bf(float f) {
  unsigned int u = __builtin_bit_cast(unsigned int, f);
  u += 0x7FFFu + ((u >> 16) & 1u);          // round-to-nearest-even
  return (unsigned short)(u >> 16);
}

__global__ __launch_bounds__(256) void cvt_f32_bf16(
    unsigned short* __restrict__ dst, const float* __restrict__ src, int n) {
  int i = blockIdx.x * 256 + threadIdx.x;
  if (i < n) dst[i] = f2bf(src[i]);
}

// ------------------------------------------------------------- WMMA GEMM
// C(M,N) f32 = A(M,K)bf16 * B(N,K)bf16^T   (both operands K-major)
// block = 256 threads = 8 waves; block tile 128(M) x 64(N); wave owns 16 rows.
// B tile (64 x 32 bf16 = 4KB) staged in LDS via async copies (ASYNCcnt),
// double buffered: stage s+1 in flight while WMMAs consume stage s.
// mode 0: plain store. mode 1: C = softplus(C + bias[n])   (delta path)
__global__ __launch_bounds__(256) void wmma_gemm_bf16(
    float* __restrict__ C,
    const unsigned short* __restrict__ A,
    const unsigned short* __restrict__ B,
    const float* __restrict__ bias,
    int M, int N, int K, int lda, int ldb, int ldc, int mode)
{
  __shared__ __align__(64) unsigned short sB[2][64 * KSTEP];   // 2 x 4KB

  const int tid   = threadIdx.x;
  const int lane  = tid & 31;
  const int wave  = tid >> 5;
  const int mTile = blockIdx.y * 128 + wave * 16;
  const int nTile = blockIdx.x * 64;
  const int r  = lane & 15;   // row-in-tile (A) / column (B,C)
  const int hi = lane >> 4;   // lane half selects K sub-block

  // ---- async staging assignment: thread copies one 16B segment of B tile
  const int srow = tid >> 2;                       // 0..63 (row within n-tile)
  const int sseg = tid & 3;                        // 0..3 (16B segment in k)
  const unsigned short* gB = B + (size_t)(nTile + srow) * ldb + sseg * 8;
  const unsigned ldsElem = (unsigned)(srow * KSTEP + sseg * 8);
  const unsigned lds0 = (unsigned)(size_t)&sB[0][ldsElem];
  const unsigned lds1 = (unsigned)(size_t)&sB[1][ldsElem];

  const int nk = K / KSTEP;

  // prologue: issue stage 0 (and stage 1) async copies
  asm volatile("global_load_async_to_lds_b128 %0, %1, off"
               :: "v"(lds0), "v"(gB) : "memory");
  if (nk > 1) {
    const unsigned short* g1 = gB + KSTEP;
    asm volatile("global_load_async_to_lds_b128 %0, %1, off"
                 :: "v"(lds1), "v"(g1) : "memory");
  }

  v8f acc0 = {}, acc1 = {}, acc2 = {}, acc3 = {};
  const unsigned short* Arow = A + (size_t)(mTile + r) * lda;

  for (int ks = 0; ks < nk; ++ks) {
    const int k0 = ks * KSTEP;

    // wait for the current stage's async copy (1 outstanding = next stage)
    if (ks < nk - 1) asm volatile("s_wait_asynccnt 0x1" ::: "memory");
    else             asm volatile("s_wait_asynccnt 0x0" ::: "memory");
    __syncthreads();

    const unsigned short* sb = &sB[ks & 1][0];

    __builtin_prefetch(Arow + k0 + 256, 0, 0);   // gfx1250 global_prefetch_b8

    // A fragment 16x32: lane half 0 holds K 0..15, half 1 16..31
    ushort8 a0 = *(const ushort8*)(Arow + k0 + hi * 16);
    ushort8 a1 = *(const ushort8*)(Arow + k0 + hi * 16 + 8);
    ushort16 au = __builtin_shufflevector(a0, a1, 0,1,2,3,4,5,6,7,
                                                  8,9,10,11,12,13,14,15);
    v16bf af = __builtin_bit_cast(v16bf, au);

    // B fragments 32x16 from LDS: lane = column, half 0 -> K 0..15, half 1 -> 16..31
    v16bf bf0 = *(const v16bf*)(sb + ( 0 + r) * KSTEP + hi * 16);
    v16bf bf1 = *(const v16bf*)(sb + (16 + r) * KSTEP + hi * 16);
    v16bf bf2 = *(const v16bf*)(sb + (32 + r) * KSTEP + hi * 16);
    v16bf bf3 = *(const v16bf*)(sb + (48 + r) * KSTEP + hi * 16);

    acc0 = __builtin_amdgcn_wmma_f32_16x16x32_bf16(false, af, false, bf0, (short)0, acc0, false, false);
    acc1 = __builtin_amdgcn_wmma_f32_16x16x32_bf16(false, af, false, bf1, (short)0, acc1, false, false);
    acc2 = __builtin_amdgcn_wmma_f32_16x16x32_bf16(false, af, false, bf2, (short)0, acc2, false, false);
    acc3 = __builtin_amdgcn_wmma_f32_16x16x32_bf16(false, af, false, bf3, (short)0, acc3, false, false);

    __syncthreads();   // all waves done reading buf[ks&1] before it is refilled

    if (ks + 2 < nk) {
      const unsigned short* gn = gB + (ks + 2) * KSTEP;
      const unsigned ldsn = (ks & 1) ? lds1 : lds0;
      asm volatile("global_load_async_to_lds_b128 %0, %1, off"
                   :: "v"(ldsn), "v"(gn) : "memory");
    }
  }

  // C/D layout: VGPR i -> row mTile + i + hi*8, column nTile + t*16 + r
  const int rowBase = mTile + hi * 8;
  v8f accs[4] = {acc0, acc1, acc2, acc3};
  #pragma unroll
  for (int t = 0; t < 4; ++t) {
    const int n = nTile + t * 16 + r;
    float bv = (mode == 1) ? bias[n] : 0.0f;
    #pragma unroll
    for (int i = 0; i < 8; ++i) {
      float v = accs[t][i];
      if (mode == 1) {               // softplus(v + b_dt)
        v += bv;
        v = (v > 20.0f) ? v : log1pf(__expf(v));
      }
      C[(size_t)(rowBase + i) * ldc + n] = v;
    }
  }
}

// ------------------------------------------ causal depthwise conv + SiLU
// u[m,d] = silu( sum_{j=0..3} cw[d,j]*xz_u[m-3+j, d] + cb[d] ), per batch seq
__global__ __launch_bounds__(256) void conv_silu_kernel(
    float* __restrict__ u, const float* __restrict__ xz,
    const float* __restrict__ cw, const float* __restrict__ cb)
{
  int idx = blockIdx.x * 256 + threadIdx.x;    // MROWS * D_INNER
  int d = idx & (D_INNER - 1);
  int m = idx >> 10;
  int s = m & (SEQ - 1);
  float acc = cb[d];
  #pragma unroll
  for (int j = 0; j < D_CONV; ++j) {
    int si = s - (D_CONV - 1) + j;
    if (si >= 0)
      acc += cw[d * D_CONV + j] * xz[(size_t)(m - (D_CONV - 1) + j) * (2 * D_INNER) + d];
  }
  u[idx] = acc / (1.0f + __expf(-acc));        // silu
}

// -------------------------------------- selective scan fused with gating
// one block = (batch b, 256 channels); thread owns channel d, 16-state h in regs
// per step: B/C (32 floats, shared by all channels) broadcast via LDS
__global__ __launch_bounds__(256) void scan_gate_kernel(
    unsigned short* __restrict__ yz,        // bf16 out: y * silu(z)
    const float* __restrict__ delta,
    const float* __restrict__ u,
    const float* __restrict__ xdbl,         // cols 32..47 = B, 48..63 = C
    const float* __restrict__ xz,           // cols 1024..2047 = z
    const float* __restrict__ A_log,
    const float* __restrict__ Dvec)
{
  const int b = blockIdx.x >> 2;
  const int d = ((blockIdx.x & 3) << 8) + threadIdx.x;

  float Acoef[D_STATE];
  #pragma unroll
  for (int n = 0; n < D_STATE; ++n) Acoef[n] = -__expf(A_log[d * D_STATE + n]);
  const float Dv = Dvec[d];

  float h[D_STATE];
  #pragma unroll
  for (int n = 0; n < D_STATE; ++n) h[n] = 0.0f;

  __shared__ float bc[2 * D_STATE];

  for (int s = 0; s < SEQ; ++s) {
    const size_t m = (size_t)b * SEQ + s;
    if (threadIdx.x < 2 * D_STATE)
      bc[threadIdx.x] = xdbl[m * 64 + DT_RANK + threadIdx.x];
    __syncthreads();

    const float dv = delta[m * D_INNER + d];
    const float uv = u[m * D_INNER + d];
    const float dvu = dv * uv;
    float y = 0.0f;
    #pragma unroll
    for (int n = 0; n < D_STATE; ++n) {
      float dA = __expf(dv * Acoef[n]);
      h[n] = dA * h[n] + dvu * bc[n];
      y += h[n] * bc[D_STATE + n];
    }
    y += uv * Dv;

    const float z = xz[m * (2 * D_INNER) + D_INNER + d];
    const float g = z / (1.0f + __expf(-z));
    yz[m * D_INNER + d] = f2bf(y * g);
    __syncthreads();
  }
}

// ---------------------------------------------------------------- launcher
extern "C" void kernel_launch(void* const* d_in, const int* in_sizes, int n_in,
                              void* d_out, int out_size, void* d_ws, size_t ws_size,
                              hipStream_t stream) {
  const float* x     = (const float*)d_in[0];
  const float* W_in  = (const float*)d_in[1];
  const float* cw    = (const float*)d_in[2];
  const float* cb    = (const float*)d_in[3];
  const float* W_x   = (const float*)d_in[4];
  const float* W_dt  = (const float*)d_in[5];
  const float* b_dt  = (const float*)d_in[6];
  const float* A_log = (const float*)d_in[7];
  const float* Dvec  = (const float*)d_in[8];
  const float* W_out = (const float*)d_in[9];
  float* out = (float*)d_out;

  // ---- workspace carve-up (256B aligned) ----
  char* ws = (char*)d_ws;
  size_t off = 0;
  auto carve = [&](size_t bytes) -> char* {
    char* p = ws + off;
    off += (bytes + 255) & ~(size_t)255;
    return p;
  };
  float*          xz       = (float*)         carve((size_t)MROWS * 2 * D_INNER * 4);
  float*          u        = (float*)         carve((size_t)MROWS * D_INNER * 4);
  unsigned short* u_bf     = (unsigned short*)carve((size_t)MROWS * D_INNER * 2);
  float*          xdbl     = (float*)         carve((size_t)MROWS * 64 * 4);
  unsigned short* xdbl_bf  = (unsigned short*)carve((size_t)MROWS * 64 * 2);
  float*          delta    = (float*)         carve((size_t)MROWS * D_INNER * 4);
  unsigned short* yz_bf    = (unsigned short*)carve((size_t)MROWS * D_INNER * 2);
  float*          xout     = (float*)         carve((size_t)MROWS * DIM * 4);
  unsigned short* x_bf     = (unsigned short*)carve((size_t)MROWS * DIM * 2);
  unsigned short* w_in_bf  = (unsigned short*)carve((size_t)2 * D_INNER * DIM * 2);
  unsigned short* w_x_bf   = (unsigned short*)carve((size_t)64 * D_INNER * 2);
  unsigned short* w_dt_bf  = (unsigned short*)carve((size_t)D_INNER * DT_RANK * 2);
  unsigned short* w_out_bf = (unsigned short*)carve((size_t)DIM * D_INNER * 2);

  auto cvt = [&](unsigned short* dst, const float* src, int n) {
    cvt_f32_bf16<<<(n + 255) / 256, 256, 0, stream>>>(dst, src, n);
  };
  auto gemm = [&](float* C, const unsigned short* A, const unsigned short* B,
                  const float* bias, int M, int N, int K,
                  int lda, int ldb, int ldc, int mode) {
    dim3 grid(N / 64, M / 128);
    wmma_gemm_bf16<<<grid, 256, 0, stream>>>(C, A, B, bias, M, N, K, lda, ldb, ldc, mode);
  };

  for (int layer = 0; layer < N_LAYERS; ++layer) {
    const float* xin = (layer == 0) ? x : xout;
    float* dst = (layer == N_LAYERS - 1) ? out : xout;

    const float* Wi  = W_in  + (size_t)layer * 2 * D_INNER * DIM;
    const float* cwl = cw    + (size_t)layer * D_INNER * D_CONV;
    const float* cbl = cb    + (size_t)layer * D_INNER;
    const float* Wxl = W_x   + (size_t)layer * 64 * D_INNER;
    const float* Wdl = W_dt  + (size_t)layer * D_INNER * DT_RANK;
    const float* bdl = b_dt  + (size_t)layer * D_INNER;
    const float* Al  = A_log + (size_t)layer * D_INNER * D_STATE;
    const float* Dl  = Dvec  + (size_t)layer * D_INNER;
    const float* Wol = W_out + (size_t)layer * DIM * D_INNER;

    // convert this layer's operands to bf16
    cvt(x_bf,     xin, MROWS * DIM);
    cvt(w_in_bf,  Wi,  2 * D_INNER * DIM);
    cvt(w_x_bf,   Wxl, 64 * D_INNER);
    cvt(w_dt_bf,  Wdl, D_INNER * DT_RANK);
    cvt(w_out_bf, Wol, DIM * D_INNER);

    // xz = x @ W_in^T         (4096 x 2048 x 512)
    gemm(xz, x_bf, w_in_bf, nullptr, MROWS, 2 * D_INNER, DIM, DIM, DIM, 2 * D_INNER, 0);

    // u = silu(causal_conv(xz[:, :1024]))
    conv_silu_kernel<<<(MROWS * D_INNER) / 256, 256, 0, stream>>>(u, xz, cwl, cbl);
    cvt(u_bf, u, MROWS * D_INNER);

    // xdbl = u @ W_x^T        (4096 x 64 x 1024)
    gemm(xdbl, u_bf, w_x_bf, nullptr, MROWS, 64, D_INNER, D_INNER, D_INNER, 64, 0);
    cvt(xdbl_bf, xdbl, MROWS * 64);

    // delta = softplus(dt @ W_dt^T + b_dt)  (dt = xdbl[:, :32], lda = 64)
    gemm(delta, xdbl_bf, w_dt_bf, bdl, MROWS, D_INNER, DT_RANK, 64, DT_RANK, D_INNER, 1);

    // selective scan + silu(z) gate -> bf16
    scan_gate_kernel<<<BATCH * (D_INNER / 256), 256, 0, stream>>>(
        yz_bf, delta, u, xdbl, xz, Al, Dl);

    // out = (y * silu(z)) @ W_out^T   (4096 x 512 x 1024)
    gemm(dst, yz_bf, w_out_bf, nullptr, MROWS, DIM, D_INNER, D_INNER, D_INNER, DIM, 0);
  }
}
